// image_model_22316650070872
// MI455X (gfx1250) — matrix operand order repaired and checked by reading, not compiled, not run on hardware
//
#include <hip/hip_runtime.h>
#include <hip/hip_bf16.h>

// MI455X (gfx1250) implementation.
// Roofline: e-noise traffic = 172 MB (fits 192MB L2) -> ~7us floor @23.3TB/s.
// fc1 is bandwidth-bound (16 FLOP/byte of e3) -> fuse weight-gen into GEMM.
// conv2 is compute-bound (10.2 GFLOP) -> f16 WMMA (16x16x32, f32 accum);
// its inner loop is fully LDS-resident (weights + im2col offsets + images),
// so gathers are 32-bit ds_load_u16 with no 64-bit address math.

typedef __attribute__((ext_vector_type(16))) _Float16       v16h;
typedef __attribute__((ext_vector_type(8)))  _Float16       v8h;
typedef __attribute__((ext_vector_type(8)))  float          v8f;
typedef __attribute__((ext_vector_type(8)))  unsigned short v8us;

#define ET  431080   // e row length
#define S_  100
#define B_  32

__device__ __forceinline__ float softplusf(float r) {
    float ar = fabsf(r);
    return fmaxf(r, 0.0f) + log1pf(__expf(-ar));
}

// B-fragment (16-bit, KxN = 32x16): lane holds column n = lane&15,
// K-half kb = (lane<16 ? 0 : 8); element i <-> K = kb + (i<8 ? i : i+8).
__device__ __forceinline__ v16h lds_bfrag(const _Float16* row, int off) {
    v8h lo = *(const v8h*)(row + off);
    v8h hi = *(const v8h*)(row + off + 16);
    return __builtin_shufflevector(lo, hi, 0,1,2,3,4,5,6,7,8,9,10,11,12,13,14,15);
}

// A-fragment (16-bit, MxK = 16x32): lane holds row m = lane&15, K-half
// khalf = (lane<16 ? 0 : 8); element i <-> K = khalf + (i<8 ? i : i+8).
__device__ __forceinline__ v16h lds_afrag(const _Float16* row, int off) {
    v8h lo = *(const v8h*)(row + off);
    v8h hi = *(const v8h*)(row + off + 16);
    return __builtin_shufflevector(lo, hi, 0,1,2,3,4,5,6,7,8,9,10,11,12,13,14,15);
}

// ---------------- k0: zero output (k4 accumulates atomically) ---------------
__global__ void k0_zero(float* out) {
    int t = blockIdx.x * 256 + threadIdx.x;
    if (t < 320) out[t] = 0.0f;
}

// ---------------- k1: conv1 + relu + maxpool (VALU fp32, tiny) --------------
__global__ __launch_bounds__(256) void k1_conv1(
    const float* __restrict__ x, const float* __restrict__ e,
    const float* __restrict__ mu_w, const float* __restrict__ rho_w,
    const float* __restrict__ mu_b, const float* __restrict__ rho_b,
    _Float16* __restrict__ h1)
{
    __shared__ float xs[784];
    __shared__ float ws[500];
    __shared__ float bs[20];
    const int s = blockIdx.x;            // 100
    const int b = blockIdx.y;            // 32
    const int t = threadIdx.x;
    const float* er = e + (size_t)s * ET;          // e1 region starts at 0
    for (int i = t; i < 784; i += 256) xs[i] = x[b * 784 + i];
    for (int i = t; i < 500; i += 256)
        ws[i] = mu_w[i] + softplusf(rho_w[i]) * er[i];
    if (t < 20) bs[t] = mu_b[t] + softplusf(rho_b[t]) * er[500 + t];
    __syncthreads();
    for (int oid = t; oid < 2880; oid += 256) {    // 20ch * 12 * 12
        int c = oid / 144, rem = oid % 144;
        int py = rem / 12, px = rem % 12;
        const float* wc = ws + c * 25;
        float m = -1e30f;
        #pragma unroll
        for (int dy0 = 0; dy0 < 2; ++dy0)
        #pragma unroll
        for (int dx0 = 0; dx0 < 2; ++dx0) {
            int y0 = 2 * py + dy0, x0 = 2 * px + dx0;
            float acc = bs[c];
            #pragma unroll
            for (int ky = 0; ky < 5; ++ky)
            #pragma unroll
            for (int kx = 0; kx < 5; ++kx)
                acc = fmaf(xs[(y0 + ky) * 28 + x0 + kx], wc[ky * 5 + kx], acc);
            m = fmaxf(m, acc);
        }
        m = fmaxf(m, 0.0f);
        h1[(((size_t)b * S_ + s) * 20 + c) * 144 + rem] = (_Float16)m;
    }
}

// ------ k2: conv2 as implicit-im2col WMMA GEMM + relu + maxpool -------------
// Per sample s: C[2048 pos, 50 ch] = P[2048, 512pad] * W[512pad, 50].
// W (from e2), the 4 input images, and the im2col offset table are all staged
// into LDS (76.3 KB of the 320 KB WGP LDS); the WMMA loop is LDS-only.
__global__ __launch_bounds__(256) void k2_conv2(
    const _Float16* __restrict__ h1, const float* __restrict__ e,
    const float* __restrict__ mu_w, const float* __restrict__ rho_w,
    const float* __restrict__ mu_b, const float* __restrict__ rho_b,
    _Float16* __restrict__ h2)
{
    __shared__ _Float16       Wl[50][520];  // [outc][k], stride -> bank-spread
    __shared__ _Float16       h1s[4][2880]; // this block's 4 images (20x12x12)
    __shared__ unsigned short offT[512];    // k -> cin*144 + dy*12 + dx
    __shared__ float          bias[64];
    const int s     = blockIdx.x;        // 100
    const int chunk = blockIdx.y;        // 8 chunks of 256 positions (4 images)
    const int t = threadIdx.x, lane = t & 31, wv = t >> 5;
    const float* ew = e + (size_t)s * ET + 520;    // e2 region
    for (int idx = t; idx < 50 * 512; idx += 256) {
        int n = idx >> 9, k = idx & 511;
        float v = 0.0f;
        if (k < 500) {
            int g = n * 500 + k;
            v = mu_w[g] + softplusf(rho_w[g]) * ew[g];
        }
        Wl[n][k] = (_Float16)v;
    }
    for (int idx = t; idx < 4 * 2880; idx += 256) {
        int img = idx / 2880, j = idx - img * 2880;
        h1s[img][j] = h1[(((size_t)(chunk * 4 + img) * S_ + s) * 20) * 144 + j];
    }
    for (int k = t; k < 512; k += 256) {
        unsigned short o = 0;
        if (k < 500) {
            int cin = k / 25, r = k % 25;
            o = (unsigned short)(cin * 144 + (r / 5) * 12 + (r % 5));
        }
        offT[k] = o;
    }
    if (t < 64) {
        float v = 0.0f;
        if (t < 50) v = mu_b[t] + softplusf(rho_b[t]) * ew[25000 + t];
        bias[t] = v;
    }
    __syncthreads();

    const int mlane  = lane & 15;
    const int khalfA = (lane & 16) ? 8 : 0;   // A: K 0-7/16-23 vs 8-15/24-31
    const int kb     = (lane & 16) ? 8 : 0;   // B: K 0-7/16-23 vs 8-15/24-31
    const _Float16* rows[4];
    rows[0] = Wl[mlane];
    rows[1] = Wl[16 + mlane];
    rows[2] = Wl[32 + mlane];
    rows[3] = Wl[(48 + mlane) < 50 ? (48 + mlane) : 49];  // clamp; cols>=50 unused

    #pragma unroll
    for (int mt = 0; mt < 2; ++mt) {
        const int tileBase = chunk * 256 + (wv * 2 + mt) * 16;   // 16-pos tile
        const int posAll = tileBase + mlane;                     // this lane's row
        const int bimg = posAll >> 6;                            // global image
        const int rem  = posAll & 63;
        const int posOff = (rem >> 3) * 12 + (rem & 7);          // yy*12 + xx
        const _Float16* h1pos = &h1s[(posAll >> 6) & 3][posOff]; // LDS, 32-bit addr
        v8f acc[4] = { {}, {}, {}, {} };
        // fast path: ks 0..14 -> all k < 500, pure table-lookup LDS gathers
        for (int ks = 0; ks < 15; ++ks) {
            const int k0 = ks * 32 + khalfA;
            v8us o0 = *(const v8us*)&offT[k0];
            v8us o1 = *(const v8us*)&offT[k0 + 16];
            v16h a;
            #pragma unroll
            for (int i = 0; i < 8; ++i) a[i]     = h1pos[o0[i]];
            #pragma unroll
            for (int i = 0; i < 8; ++i) a[8 + i] = h1pos[o1[i]];
            const int off = ks * 32 + kb;
            #pragma unroll
            for (int nt = 0; nt < 4; ++nt) {
                v16h bf = lds_bfrag(rows[nt], off);
                acc[nt] = __builtin_amdgcn_wmma_f32_16x16x32_f16(
                    false, a, false, bf, (short)0, acc[nt], false, false);
            }
        }
        { // peeled ks = 15: k in [480,512), zero-fill k >= 500
            v16h a;
            #pragma unroll
            for (int i = 0; i < 16; ++i) {
                int k = 480 + khalfA + ((i < 8) ? i : (i + 8));
                float v = 0.0f;
                if (k < 500) v = (float)h1pos[offT[k]];
                a[i] = (_Float16)v;
            }
            const int off = 480 + kb;
            #pragma unroll
            for (int nt = 0; nt < 4; ++nt) {
                v16h bf = lds_bfrag(rows[nt], off);
                acc[nt] = __builtin_amdgcn_wmma_f32_16x16x32_f16(
                    false, a, false, bf, (short)0, acc[nt], false, false);
            }
        }
        // epilogue: bias + relu + 2x2 maxpool. C layout: elem r -> M=r(+8), N=lane&15.
        const int y0 = (tileBase & 63) >> 3;  // even
        const int py = y0 >> 1;
        const size_t obase = ((size_t)bimg * S_ + s) * 800;
        #pragma unroll
        for (int nt = 0; nt < 4; ++nt) {
            int ch = nt * 16 + mlane;
            float bsv = bias[ch];
            float p[4];
            #pragma unroll
            for (int px = 0; px < 4; ++px) {
                float v0 = fmaxf(acc[nt][2 * px]     + bsv, 0.0f);
                float v1 = fmaxf(acc[nt][2 * px + 1] + bsv, 0.0f);
                p[px] = fmaxf(v0, v1);       // max over x pair (row y0 or y0+1)
            }
            #pragma unroll
            for (int px = 0; px < 4; ++px) {
                float o2 = __shfl_xor(p[px], 16, 32);   // pair y0 with y0+1
                p[px] = fmaxf(p[px], o2);
                if (lane < 16 && ch < 50)
                    h2[obase + ch * 16 + py * 4 + px] = (_Float16)p[px];
            }
        }
    }
}

// ---------- k3: fc1, weight-gen fused into WMMA GEMM (HBM-bound) ------------
// Per (s, 64-output chunk): C[32,64] = H2[32,800] x W^T tile.  H2 is staged
// once into LDS (conflict-free stride); W is staged per K-step straight from
// e3 (coalesced, prefetched) and converted to f16.
__global__ __launch_bounds__(256) void k3_fc1(
    const _Float16* __restrict__ h2, const float* __restrict__ e,
    const float* __restrict__ mu_w, const float* __restrict__ rho_w,
    const float* __restrict__ mu_b, const float* __restrict__ rho_b,
    float* __restrict__ h3)
{
    __shared__ _Float16 h2s[32][808];    // 51.7 KB, 404 dw stride -> no conflicts
    __shared__ _Float16 Wl[64][40];      // [n][k] 32-wide + pad
    const int s  = blockIdx.x;           // 100
    const int nb = blockIdx.y;           // 8 -> 64 outputs each (500 padded 512)
    const int t = threadIdx.x, lane = t & 31, wv = t >> 5;
    const int mt = wv >> 2, nt = wv & 3; // 2 M-tiles x 4 N-tiles
    const int mlane  = lane & 15;
    const int khalfA = (lane & 16) ? 8 : 0;
    const int kbB    = (lane & 16) ? 8 : 0;
    const int nBase  = nb * 64;
    const float* ew = e + (size_t)s * ET + 25570;  // e3 region

    for (int idx = t; idx < 32 * 800; idx += 256) {
        int b = idx / 800, k = idx % 800;
        h2s[b][k] = h2[((size_t)b * S_ + s) * 800 + k];
    }
    const _Float16* arow = h2s[mt * 16 + mlane];

    v8f acc = {};
    for (int ks = 0; ks < 25; ++ks) {
        __syncthreads();                 // first pass also covers h2s staging
        for (int idx = t; idx < 64 * 32; idx += 256) {
            int n = idx >> 5, k = idx & 31;
            int o = nBase + n;
            float v = 0.0f;
            if (o < 500) {
                size_t g = (size_t)o * 800 + ks * 32 + k;
                v = mu_w[g] + softplusf(rho_w[g]) * ew[g];
                __builtin_prefetch(&ew[g + 32], 0, 1);   // next K-step of e3
            }
            Wl[n][k] = (_Float16)v;
        }
        __syncthreads();
        v16h a  = lds_afrag(arow, ks * 32 + khalfA);
        v16h bf = lds_bfrag(Wl[nt * 16 + mlane], kbB);
        acc = __builtin_amdgcn_wmma_f32_16x16x32_f16(
            false, a, false, bf, (short)0, acc, false, false);
    }
    // epilogue: bias + relu -> h3 fp32
    const int o = nBase + nt * 16 + mlane;
    float bsv = 0.0f;
    if (o < 500) bsv = mu_b[o] + softplusf(rho_b[o]) * ew[400000 + o];
    #pragma unroll
    for (int r = 0; r < 8; ++r) {
        int bi = mt * 16 + r + ((lane & 16) ? 8 : 0);
        float v = fmaxf(acc[r] + bsv, 0.0f);
        if (o < 500) h3[((size_t)bi * S_ + s) * 500 + o] = v;
    }
}

// ---------- k4: fc2 + log_softmax + mean over samples (tiny) ----------------
__global__ __launch_bounds__(256) void k4_fc2(
    const float* __restrict__ h3, const float* __restrict__ e,
    const float* __restrict__ mu_w, const float* __restrict__ rho_w,
    const float* __restrict__ mu_b, const float* __restrict__ rho_b,
    float* __restrict__ out)
{
    __shared__ float w2[10][500];
    __shared__ float b2[10];
    __shared__ float lg[32][10];
    const int s = blockIdx.x;            // 100
    const int t = threadIdx.x, lane = t & 31, wv = t >> 5;
    const float* ew = e + (size_t)s * ET + 426070; // e4 region
    for (int idx = t; idx < 5000; idx += 256)
        w2[idx / 500][idx % 500] = mu_w[idx] + softplusf(rho_w[idx]) * ew[idx];
    if (t < 10) b2[t] = mu_b[t] + softplusf(rho_b[t]) * ew[5000 + t];
    __syncthreads();
    #pragma unroll
    for (int j = 0; j < 4; ++j) {
        int b = wv * 4 + j;
        const float* hrow = h3 + ((size_t)b * S_ + s) * 500;
        for (int o = 0; o < 10; ++o) {
            float p = 0.0f;
            for (int i = lane; i < 500; i += 32)
                p = fmaf(hrow[i], w2[o][i], p);
            #pragma unroll
            for (int d = 16; d >= 1; d >>= 1) p += __shfl_xor(p, d, 32);
            if (lane == 0) lg[b][o] = p + b2[o];
        }
    }
    __syncthreads();
    if (t < 32) {
        float m = -1e30f;
        #pragma unroll
        for (int o = 0; o < 10; ++o) m = fmaxf(m, lg[t][o]);
        float sum = 0.0f;
        #pragma unroll
        for (int o = 0; o < 10; ++o) sum += __expf(lg[t][o] - m);
        float lse = m + __logf(sum);
        #pragma unroll
        for (int o = 0; o < 10; ++o)
            atomicAdd(&out[t * 10 + o], (lg[t][o] - lse) * 0.01f);
    }
}

extern "C" void kernel_launch(void* const* d_in, const int* in_sizes, int n_in,
                              void* d_out, int out_size, void* d_ws, size_t ws_size,
                              hipStream_t stream) {
    (void)in_sizes; (void)n_in; (void)out_size; (void)ws_size;
    const float* x     = (const float*)d_in[0];
    const float* e     = (const float*)d_in[1];
    const float* c1mw  = (const float*)d_in[2];
    const float* c1rw  = (const float*)d_in[3];
    const float* c1mb  = (const float*)d_in[4];
    const float* c1rb  = (const float*)d_in[5];
    const float* c2mw  = (const float*)d_in[6];
    const float* c2rw  = (const float*)d_in[7];
    const float* c2mb  = (const float*)d_in[8];
    const float* c2rb  = (const float*)d_in[9];
    const float* a1mw  = (const float*)d_in[10];
    const float* a1rw  = (const float*)d_in[11];
    const float* a1mb  = (const float*)d_in[12];
    const float* a1rb  = (const float*)d_in[13];
    const float* a2mw  = (const float*)d_in[14];
    const float* a2rw  = (const float*)d_in[15];
    const float* a2mb  = (const float*)d_in[16];
    const float* a2rb  = (const float*)d_in[17];
    float* out = (float*)d_out;

    // workspace: h1 f16 (18,432,000 B) | h2 f16 (5,120,000 B) | h3 f32 (6,400,000 B)
    _Float16* h1 = (_Float16*)d_ws;
    _Float16* h2 = (_Float16*)((char*)d_ws + 18432000);
    float*    h3 = (float*)   ((char*)d_ws + 23552000);

    k0_zero <<<2, 256, 0, stream>>>(out);
    k1_conv1<<<dim3(100, 32), 256, 0, stream>>>(x,  e, c1mw, c1rw, c1mb, c1rb, h1);
    k2_conv2<<<dim3(100, 8),  256, 0, stream>>>(h1, e, c2mw, c2rw, c2mb, c2rb, h2);
    k3_fc1  <<<dim3(100, 8),  256, 0, stream>>>(h2, e, a1mw, a1rw, a1mb, a1rb, h3);
    k4_fc2  <<<100,           256, 0, stream>>>(h3, e, a2mw, a2rw, a2mb, a2rb, out);
}
